// RWKV_Tmix_x070_31215822307714
// MI455X (gfx1250) — matrix operand order repaired and checked by reading, not compiled, not run on hardware
//
#include <hip/hip_runtime.h>

// ---------------------------------------------------------------------------
// RWKV7 time-mix forward for MI455X (gfx1250, wave32, WMMA).
// B=2, T=2048, C=2048, H=32, N=64, M=B*T=4096.
// ---------------------------------------------------------------------------

#define B_ 2
#define T_ 2048
#define C_ 2048
#define H_ 32
#define N_ 64
#define M_ (B_ * T_)

typedef __attribute__((ext_vector_type(16))) __bf16 v16bf;
typedef __attribute__((ext_vector_type(8)))  __bf16 v8bf;
typedef __attribute__((ext_vector_type(4)))  __bf16 v4bf;
typedef __attribute__((ext_vector_type(8)))  float  v8f;
typedef __attribute__((ext_vector_type(4)))  float  v4f;

__device__ __forceinline__ float sigmoidf_(float x) { return 1.0f / (1.0f + __expf(-x)); }
__device__ __forceinline__ float bfrt(float x) { return (float)(__bf16)x; }  // bf16 round-trip

__device__ __forceinline__ v8f vzero8() {
    v8f z = {0.f, 0.f, 0.f, 0.f, 0.f, 0.f, 0.f, 0.f};
    return z;
}

__device__ __forceinline__ v16bf cat16(v8bf lo, v8bf hi) {
    return __builtin_shufflevector(lo, hi, 0, 1, 2, 3, 4, 5, 6, 7,
                                   8, 9, 10, 11, 12, 13, 14, 15);
}

// ---------------------------------------------------------------------------
// fp32 -> bf16 conversion for weights
// ---------------------------------------------------------------------------
__global__ void cvt_bf16_kernel(const float* __restrict__ in, __bf16* __restrict__ out, int n) {
    int i = blockIdx.x * blockDim.x + threadIdx.x;
    if (i < n) out[i] = (__bf16)in[i];
}

// ---------------------------------------------------------------------------
// Unified WMMA GEMM:  out[M,Nn] = epilogue( A[M,K] @ Wb[K,Nn] )
// AMODE: 0 = A from Abuf; 1 = token-shift mix of x; 2 = Abuf * Abuf2 (y*g)
// EMODE: 0 none | 1 tanh | 2 sigmoid | 3 w-decay | 4 sigmoid(bias+acc) | 5 v-lerp
// Block: 256 threads = 8 waves; block tile 128(M) x 64(N); K step 32.
// Each wave computes a 32x32 tile via 4 x v_wmma_f32_16x16x32_bf16.
// ---------------------------------------------------------------------------
template <int AMODE, int EMODE>
__global__ __launch_bounds__(256)
void wmma_gemm(const float* __restrict__ Abuf,
               const float* __restrict__ Abuf2,
               const float* __restrict__ xshift,
               const float* __restrict__ mix,
               const __bf16* __restrict__ Wb,
               const float* __restrict__ bias,
               const float* __restrict__ vfirst,
               float* __restrict__ vbuf,
               float* __restrict__ out,
               int M, int K, int Nn, int T) {
    // A tile row-major [m][k]; B tile TRANSPOSED in LDS: [n][k] so each lane's
    // fragment is two contiguous 8-half runs (two ds_load_b128).
    __shared__ __align__(16) __bf16 Atile[128][32];
    __shared__ __align__(16) __bf16 Btile[64][32];

    const int tid = threadIdx.x;
    const int m0 = blockIdx.x * 128;
    const int n0 = blockIdx.y * 64;

    v8f acc00 = vzero8(), acc01 = vzero8(), acc10 = vzero8(), acc11 = vzero8();

    const int wave  = tid >> 5;
    const int lane  = tid & 31;
    const int wm    = (wave >> 1) * 32;   // wave M offset in block tile
    const int wn    = (wave & 1) * 32;    // wave N offset
    const int lhalf = lane >> 4;
    const int l16   = lane & 15;

    for (int kt = 0; kt < K; kt += 32) {
        // ---- cooperative A tile load (128x32), vectorized + fused prologue ----
        // each thread: 4 quads of 4 contiguous k -> global_load_b128 + ds_store_b64
#pragma unroll
        for (int i = 0; i < 4; ++i) {
            int idx = i * 256 + tid;          // 0..1023 quads
            int row = idx >> 3, quad = idx & 7;
            int m = m0 + row;
            int k = kt + quad * 4;
            v4f av = *(const v4f*)(Abuf + (size_t)m * K + k);
            if (AMODE == 1) {
                // token shift: prev row (or shift_state at batch start)
                bool first = ((m % T) == 0);
                const float* psrc = first ? (xshift + (size_t)(m / T) * K)
                                          : (Abuf + (size_t)(m - 1) * K);
                v4f px = *(const v4f*)(psrc + k);
                v4f mx = *(const v4f*)(mix + k);
                av = av + (px - av) * mx;
            } else if (AMODE == 2) {
                v4f gv = *(const v4f*)(Abuf2 + (size_t)m * K + k);
                av = av * gv;
            }
            v4bf o;
            o[0] = (__bf16)av[0]; o[1] = (__bf16)av[1];
            o[2] = (__bf16)av[2]; o[3] = (__bf16)av[3];
            *(v4bf*)(&Atile[row][quad * 4]) = o;
        }
        // ---- cooperative B tile load (32x64 mem -> [n][k] LDS), one b128/thread ----
        {
            int krow = tid >> 3, oct = tid & 7;   // 32 k-rows x 8 octets
            int nb = n0 + oct * 8;
            v8bf bv;
            if (nb < Nn) {                         // Nn is a multiple of 32
                bv = *(const v8bf*)(Wb + (size_t)(kt + krow) * Nn + nb);
            } else {
#pragma unroll
                for (int e = 0; e < 8; ++e) bv[e] = (__bf16)0.0f;
            }
#pragma unroll
            for (int e = 0; e < 8; ++e) Btile[oct * 8 + e][krow] = bv[e];
        }
        if (lane == 0 && kt + 32 < K)
            __builtin_prefetch(&Wb[(size_t)(kt + 32) * Nn + n0], 0, 1);
        __syncthreads();

        // ---- gather fragments (vectorized LDS reads) ----
        // A (16-bit 16x32): lane m = l16(+16), runs at halves [8*lhalf, +8) and
        // [16+8*lhalf, +8).
        const v8bf* ar0 = (const v8bf*)(&Atile[wm + l16][0]);
        const v8bf* ar1 = (const v8bf*)(&Atile[wm + 16 + l16][0]);
        v16bf af0 = cat16(ar0[lhalf], ar0[2 + lhalf]);
        v16bf af1 = cat16(ar1[lhalf], ar1[2 + lhalf]);
        // B (16-bit 32x16): lane n = l16(+16), runs at halves [8*lhalf, +8) and
        // [16+8*lhalf, +8).
        const v8bf* br0 = (const v8bf*)(&Btile[wn + l16][0]);
        const v8bf* br1 = (const v8bf*)(&Btile[wn + 16 + l16][0]);
        v16bf bf0 = cat16(br0[lhalf], br0[2 + lhalf]);
        v16bf bf1 = cat16(br1[lhalf], br1[2 + lhalf]);

        acc00 = __builtin_amdgcn_wmma_f32_16x16x32_bf16(false, af0, false, bf0, (short)0, acc00, false, false);
        acc01 = __builtin_amdgcn_wmma_f32_16x16x32_bf16(false, af0, false, bf1, (short)0, acc01, false, false);
        acc10 = __builtin_amdgcn_wmma_f32_16x16x32_bf16(false, af1, false, bf0, (short)0, acc10, false, false);
        acc11 = __builtin_amdgcn_wmma_f32_16x16x32_bf16(false, af1, false, bf1, (short)0, acc11, false, false);
        __syncthreads();
    }

    // ---- epilogue + store (C/D layout: VGPR r -> m = r + 8*half, n = lane&15) ----
#pragma unroll
    for (int ms = 0; ms < 2; ++ms) {
#pragma unroll
        for (int ns = 0; ns < 2; ++ns) {
            v8f a = (ms == 0) ? (ns == 0 ? acc00 : acc01) : (ns == 0 ? acc10 : acc11);
            int n = n0 + wn + ns * 16 + l16;
            if (n < Nn) {
                float bn = (EMODE >= 3) ? bias[n] : 0.0f;
#pragma unroll
                for (int r = 0; r < 8; ++r) {
                    int m = m0 + wm + ms * 16 + r + 8 * lhalf;
                    float v = a[r];
                    if (EMODE == 1) {
                        v = tanhf(v);
                    } else if (EMODE == 2) {
                        v = sigmoidf_(v);
                    } else if (EMODE == 3) {
                        // -softplus(-(w0+acc)) - 0.5
                        v = -log1pf(__expf(-(bn + v))) - 0.5f;
                    } else if (EMODE == 4) {
                        v = sigmoidf_(bn + v);
                    } else if (EMODE == 5) {
                        float t = sigmoidf_(bn + v);
                        float vv = vbuf[(size_t)m * Nn + n];
                        v = vv + (vfirst[(size_t)m * Nn + n] - vv) * t;
                    }
                    out[(size_t)m * Nn + n] = v;
                }
            }
        }
    }
}

// ---------------------------------------------------------------------------
// kk = normalize(k * k_k) per head; k *= (1 + (a-1)*k_a)   (one warp per b,t,h)
// ---------------------------------------------------------------------------
__global__ __launch_bounds__(256)
void kk_kernel(float* __restrict__ Kb, const float* __restrict__ Ab,
               const float* __restrict__ kkw, const float* __restrict__ kaw,
               float* __restrict__ KK, int MH, int H) {
    int g = blockIdx.x * blockDim.y + threadIdx.y;   // over M*H
    if (g >= MH) return;
    int l = threadIdx.x;                              // 0..31
    int h = g % H;
    size_t base = (size_t)g * 64;
    int c0 = h * 64 + l, c1 = c0 + 32;
    float k0 = Kb[base + l], k1 = Kb[base + 32 + l];
    float q0 = k0 * kkw[c0], q1 = k1 * kkw[c1];
    float ss = q0 * q0 + q1 * q1;
    for (int o = 16; o; o >>= 1) ss += __shfl_xor(ss, o, 32);
    float inv = 1.0f / fmaxf(sqrtf(ss), 1e-12f);
    KK[base + l] = q0 * inv;
    KK[base + 32 + l] = q1 * inv;
    float a0 = Ab[base + l], a1 = Ab[base + 32 + l];
    Kb[base + l]      = k0 * (1.0f + (a0 - 1.0f) * kaw[c0]);
    Kb[base + 32 + l] = k1 * (1.0f + (a1 - 1.0f) * kaw[c1]);
}

// ---------------------------------------------------------------------------
// RWKV7 delta-rule scan. grid = B*H blocks, 64 threads. Thread j owns S[:, j].
// ---------------------------------------------------------------------------
__global__ __launch_bounds__(64)
void scan_kernel(const float* __restrict__ R, const float* __restrict__ Kb,
                 const float* __restrict__ Vb, const float* __restrict__ Wd,
                 const float* __restrict__ KK, const float* __restrict__ Asig,
                 const float* __restrict__ s0, float* __restrict__ O,
                 float* __restrict__ Sout) {
    const int bh = blockIdx.x;
    const int b = bh / H_, h = bh % H_;
    const int j = threadIdx.x;

    float S[N_];
#pragma unroll
    for (int k = 0; k < N_; ++k)
        S[k] = s0[(((size_t)(b * H_ + h)) * N_ + k) * N_ + j];

    __shared__ float lr[N_], lew[N_], lk[N_], la[N_], lb[N_];

    for (int t = 0; t < T_; ++t) {
        size_t base = ((size_t)(b * T_ + t)) * C_ + (size_t)h * N_;
        __syncthreads();
        lr[j]  = bfrt(R[base + j]);
        lew[j] = __expf(bfrt(Wd[base + j]));
        lk[j]  = bfrt(Kb[base + j]);
        float kkj = KK[base + j];
        la[j]  = bfrt(-kkj);                      // a_t = bf16(-kk)
        lb[j]  = bfrt(kkj * Asig[base + j]);      // b_t = bf16(kk * a)
        float vj = bfrt(Vb[base + j]);
        __syncthreads();

        float sa = 0.f;
#pragma unroll
        for (int k = 0; k < N_; ++k) sa += la[k] * S[k];
        float o = 0.f;
#pragma unroll
        for (int k = 0; k < N_; ++k) {
            S[k] = S[k] * lew[k] + lb[k] * sa + lk[k] * vj;
            o += lr[k] * S[k];
        }
        O[base + j] = o;
    }
#pragma unroll
    for (int k = 0; k < N_; ++k)
        Sout[(((size_t)(b * H_ + h)) * N_ + k) * N_ + j] = S[k];
}

// ---------------------------------------------------------------------------
// GroupNorm(H groups) + (r*k*r_k).sum * v residual, in-place on O.
// ---------------------------------------------------------------------------
__global__ __launch_bounds__(256)
void gn_kernel(float* __restrict__ O, const float* __restrict__ R,
               const float* __restrict__ Kb, const float* __restrict__ Vb,
               const float* __restrict__ rk, const float* __restrict__ lnw,
               const float* __restrict__ lnb, int MH, int H) {
    int g = blockIdx.x * blockDim.y + threadIdx.y;
    if (g >= MH) return;
    int l = threadIdx.x;
    int h = g % H;
    size_t base = (size_t)g * 64;
    float o0 = O[base + l], o1 = O[base + 32 + l];
    float s = o0 + o1;
    for (int o = 16; o; o >>= 1) s += __shfl_xor(s, o, 32);
    float mu = s * (1.0f / 64.0f);
    float d0 = o0 - mu, d1 = o1 - mu;
    float v = d0 * d0 + d1 * d1;
    for (int o = 16; o; o >>= 1) v += __shfl_xor(v, o, 32);
    float rstd = rsqrtf(v * (1.0f / 64.0f) + 6.4e-4f);   // eps * 8^2
    int c0 = h * 64 + l, c1 = c0 + 32;
    float y0 = d0 * rstd * lnw[c0] + lnb[c0];
    float y1 = d1 * rstd * lnw[c1] + lnb[c1];
    float t = R[base + l] * Kb[base + l] * rk[c0] + R[base + 32 + l] * Kb[base + 32 + l] * rk[c1];
    for (int o = 16; o; o >>= 1) t += __shfl_xor(t, o, 32);
    y0 += t * Vb[base + l];
    y1 += t * Vb[base + 32 + l];
    O[base + l] = y0;
    O[base + 32 + l] = y1;
}

// ---------------------------------------------------------------------------
// copy x[:, -1, :] -> d_out tail
// ---------------------------------------------------------------------------
__global__ void xlast_kernel(const float* __restrict__ x, float* __restrict__ out, int n) {
    int i = blockIdx.x * blockDim.x + threadIdx.x;
    if (i < n) {
        int b = i / C_, c = i % C_;
        out[i] = x[((size_t)b * T_ + (T_ - 1)) * C_ + c];
    }
}

// ---------------------------------------------------------------------------
// host side
// ---------------------------------------------------------------------------
static void launch_cvt(const float* src, __bf16* dst, size_t n, hipStream_t s) {
    cvt_bf16_kernel<<<(unsigned)((n + 255) / 256), 256, 0, s>>>(src, dst, (int)n);
}

template <int AMODE, int EMODE>
static void launch_gemm(const float* A, const float* A2, const float* shift, const float* mix,
                        const __bf16* Wb, const float* bias, const float* vfirst, float* vbuf,
                        float* out, int M, int K, int Nn, hipStream_t s) {
    dim3 grid(M / 128, (Nn + 63) / 64), blk(256);
    wmma_gemm<AMODE, EMODE><<<grid, blk, 0, s>>>(A, A2, shift, mix, Wb, bias, vfirst, vbuf,
                                                 out, M, K, Nn, T_);
}

extern "C" void kernel_launch(void* const* d_in, const int* in_sizes, int n_in,
                              void* d_out, int out_size, void* d_ws, size_t ws_size,
                              hipStream_t stream) {
    const float* x        = (const float*)d_in[0];
    const float* shift    = (const float*)d_in[1];
    const float* wkv0     = (const float*)d_in[2];
    const float* v_first  = (const float*)d_in[3];
    const float* x_r      = (const float*)d_in[4];
    const float* x_w      = (const float*)d_in[5];
    const float* x_k      = (const float*)d_in[6];
    const float* x_v      = (const float*)d_in[7];
    const float* x_a      = (const float*)d_in[8];
    const float* x_g      = (const float*)d_in[9];
    const float* w0       = (const float*)d_in[10];
    const float* w1       = (const float*)d_in[11];
    const float* w2       = (const float*)d_in[12];
    const float* a0       = (const float*)d_in[13];
    const float* a1       = (const float*)d_in[14];
    const float* a2       = (const float*)d_in[15];
    const float* v0       = (const float*)d_in[16];
    const float* v1       = (const float*)d_in[17];
    const float* v2       = (const float*)d_in[18];
    const float* g1       = (const float*)d_in[19];
    const float* g2       = (const float*)d_in[20];
    const float* k_k      = (const float*)d_in[21];
    const float* k_a      = (const float*)d_in[22];
    const float* r_k      = (const float*)d_in[23];
    const float* W_r      = (const float*)d_in[24];
    const float* W_k      = (const float*)d_in[25];
    const float* W_v      = (const float*)d_in[26];
    const float* W_o      = (const float*)d_in[27];
    const float* ln_w     = (const float*)d_in[28];
    const float* ln_b     = (const float*)d_in[29];

    float* outp = (float*)d_out;
    const size_t SZ = (size_t)M_ * C_;

    // workspace layout (fp32 region, then bf16 weights)
    float* ws = (float*)d_ws;
    float* R  = ws + 0 * SZ;
    float* Kb = ws + 1 * SZ;
    float* Vb = ws + 2 * SZ;
    float* Wd = ws + 3 * SZ;
    float* KK = ws + 4 * SZ;
    float* Ab = ws + 5 * SZ;
    float* Gb = ws + 6 * SZ;
    float* Ob = ws + 7 * SZ;
    float* Hw = ws + 8 * SZ;                       // [M,64]
    float* Ha = Hw + (size_t)M_ * 64;              // [M,64]
    float* Hv = Ha + (size_t)M_ * 64;              // [M,32]
    float* Hg = Hv + (size_t)M_ * 32;              // [M,128]
    __bf16* bf = (__bf16*)(Hg + (size_t)M_ * 128);
    __bf16* Wr_b = bf;                 bf += (size_t)C_ * C_;
    __bf16* Wk_b = bf;                 bf += (size_t)C_ * C_;
    __bf16* Wv_b = bf;                 bf += (size_t)C_ * C_;
    __bf16* Wo_b = bf;                 bf += (size_t)C_ * C_;
    __bf16* w1_b = bf;                 bf += (size_t)C_ * 64;
    __bf16* w2_b = bf;                 bf += (size_t)64 * C_;
    __bf16* a1_b = bf;                 bf += (size_t)C_ * 64;
    __bf16* a2_b = bf;                 bf += (size_t)64 * C_;
    __bf16* v1_b = bf;                 bf += (size_t)C_ * 32;
    __bf16* v2_b = bf;                 bf += (size_t)32 * C_;
    __bf16* g1_b = bf;                 bf += (size_t)C_ * 128;
    __bf16* g2_b = bf;                 bf += (size_t)128 * C_;

    // 0) weights -> bf16
    launch_cvt(W_r, Wr_b, (size_t)C_ * C_, stream);
    launch_cvt(W_k, Wk_b, (size_t)C_ * C_, stream);
    launch_cvt(W_v, Wv_b, (size_t)C_ * C_, stream);
    launch_cvt(W_o, Wo_b, (size_t)C_ * C_, stream);
    launch_cvt(w1, w1_b, (size_t)C_ * 64, stream);
    launch_cvt(w2, w2_b, (size_t)64 * C_, stream);
    launch_cvt(a1, a1_b, (size_t)C_ * 64, stream);
    launch_cvt(a2, a2_b, (size_t)64 * C_, stream);
    launch_cvt(v1, v1_b, (size_t)C_ * 32, stream);
    launch_cvt(v2, v2_b, (size_t)32 * C_, stream);
    launch_cvt(g1, g1_b, (size_t)C_ * 128, stream);
    launch_cvt(g2, g2_b, (size_t)128 * C_, stream);

    // 1) stage-1 GEMMs with fused token-shift mixing (AMODE=1)
    launch_gemm<1, 0>(x, nullptr, shift, x_r, Wr_b, nullptr, nullptr, nullptr, R,  M_, C_, C_,  stream);
    launch_gemm<1, 0>(x, nullptr, shift, x_k, Wk_b, nullptr, nullptr, nullptr, Kb, M_, C_, C_,  stream);
    launch_gemm<1, 0>(x, nullptr, shift, x_v, Wv_b, nullptr, nullptr, nullptr, Vb, M_, C_, C_,  stream);
    launch_gemm<1, 1>(x, nullptr, shift, x_w, w1_b, nullptr, nullptr, nullptr, Hw, M_, C_, 64,  stream); // tanh
    launch_gemm<1, 0>(x, nullptr, shift, x_a, a1_b, nullptr, nullptr, nullptr, Ha, M_, C_, 64,  stream);
    launch_gemm<1, 0>(x, nullptr, shift, x_v, v1_b, nullptr, nullptr, nullptr, Hv, M_, C_, 32,  stream);
    launch_gemm<1, 2>(x, nullptr, shift, x_g, g1_b, nullptr, nullptr, nullptr, Hg, M_, C_, 128, stream); // sigmoid

    // 2) stage-2 LoRA GEMMs with fused epilogues
    launch_gemm<0, 3>(Hw, nullptr, nullptr, nullptr, w2_b, w0, nullptr, nullptr, Wd, M_, 64,  C_, stream); // decay
    launch_gemm<0, 4>(Ha, nullptr, nullptr, nullptr, a2_b, a0, nullptr, nullptr, Ab, M_, 64,  C_, stream); // sigmoid a
    launch_gemm<0, 5>(Hv, nullptr, nullptr, nullptr, v2_b, v0, v_first, Vb,      Vb, M_, 32,  C_, stream); // v lerp
    launch_gemm<0, 0>(Hg, nullptr, nullptr, nullptr, g2_b, nullptr, nullptr, nullptr, Gb, M_, 128, C_, stream);

    // 3) kk normalize + k update
    {
        dim3 blk(32, 8);
        int MH = M_ * H_;
        kk_kernel<<<(MH + 7) / 8, blk, 0, stream>>>(Kb, Ab, k_k, k_a, KK, MH, H_);
    }

    // 4) sequential scan; writes o and final state (into d_out tail)
    float* S_out = outp + SZ + (size_t)B_ * C_;
    scan_kernel<<<B_ * H_, 64, 0, stream>>>(R, Kb, Vb, Wd, KK, Ab, wkv0, Ob, S_out);

    // 5) GroupNorm + residual (in-place on Ob -> y)
    {
        dim3 blk(32, 8);
        int MH = M_ * H_;
        gn_kernel<<<(MH + 7) / 8, blk, 0, stream>>>(Ob, R, Kb, Vb, r_k, ln_w, ln_b, MH, H_);
    }

    // 6) out = (y * g) @ W_o   (AMODE=2: fused elementwise product)
    launch_gemm<2, 0>(Ob, Gb, nullptr, nullptr, Wo_b, nullptr, nullptr, nullptr, outp, M_, C_, C_, stream);

    // 7) x[:, -1] -> d_out tail
    xlast_kernel<<<(B_ * C_ + 255) / 256, 256, 0, stream>>>(x, outp + SZ, B_ * C_);
}
